// PointNet_FP_Module_52080773431529
// MI455X (gfx1250) — matrix operand order repaired and checked by reading, not compiled, not run on hardware
//
#include <hip/hip_runtime.h>
#include <hip/hip_bf16.h>

typedef __bf16 v16bf __attribute__((ext_vector_type(16)));
typedef float  v8f   __attribute__((ext_vector_type(8)));

#define B_   8
#define N1_  8192
#define N2_  2048
#define C1_  128
#define C2_  256
#define K0_  384          // C1+C2
#define M0_  256          // layer0 out channels
#define M1_  256          // layer1 out channels
#define R_   (B_ * N1_)   // 65536 rows
#define BN_EPS 1e-5f

// float -> bf16 (round-to-nearest-even), packed pair into one dword
__device__ __forceinline__ uint32_t pack_bf16x2(float lo, float hi) {
    uint32_t ul = __float_as_uint(lo);
    uint32_t uh = __float_as_uint(hi);
    ul = (ul + 0x7FFFu + ((ul >> 16) & 1u)) >> 16;
    uh = (uh + 0x7FFFu + ((uh >> 16) & 1u)) >> 16;
    return ul | (uh << 16);
}

// ---------------------------------------------------------------------------
// Kernel 0: zero the stats accumulators (1024 floats)
// ---------------------------------------------------------------------------
__global__ void zero_stats_kernel(float* stats) {
    stats[blockIdx.x * 256 + threadIdx.x] = 0.0f;
}

// ---------------------------------------------------------------------------
// Convert fp32 weights -> packed bf16 (one dword = 2 elems per thread)
// ---------------------------------------------------------------------------
__global__ void cvt_f32_bf16_kernel(const float* __restrict__ in,
                                    uint32_t* __restrict__ out) {
    const size_t i = (size_t)blockIdx.x * 256 + threadIdx.x;
    out[i] = pack_bf16x2(in[2 * i], in[2 * i + 1]);
}

// ---------------------------------------------------------------------------
// Kernel 1: 3-NN + inverse-distance interpolation + concat -> bf16 A0
// xyz2 for the batch staged in LDS (24 KB) via async global->LDS (ASYNCcnt).
// A0 row = [interp(256) | points1(128)]  (R_ x 384 bf16, row major)
// ---------------------------------------------------------------------------
__global__ void knn_interp_kernel(const float* __restrict__ xyz1,
                                  const float* __restrict__ xyz2,
                                  const float* __restrict__ points1,
                                  const float* __restrict__ points2,
                                  uint16_t* __restrict__ A0) {
    __shared__ float s_xyz[N2_ * 3];          // 24576 bytes
    const int tid   = threadIdx.x;
    const int b     = blockIdx.x >> 5;        // 32 chunks of 256 per batch
    const int chunk = blockIdx.x & 31;

    // async copy of this batch's xyz2 into LDS: 6 x (256 lanes * 16B) = 24 KB
    {
        const uint32_t lds_base = (uint32_t)(uintptr_t)(&s_xyz[0]);
        const uint64_t gbase    = (uint64_t)(uintptr_t)(xyz2 + (size_t)b * N2_ * 3);
        #pragma unroll
        for (int i = 0; i < 6; ++i) {
            const uint32_t off  = (uint32_t)tid * 16u + (uint32_t)i * 4096u;
            const uint32_t ldsa = lds_base + off;
            asm volatile("global_load_async_to_lds_b128 %0, %1, %2"
                         :: "v"(ldsa), "v"(off), "s"(gbase) : "memory");
        }
        asm volatile("s_wait_asynccnt 0" ::: "memory");
    }
    __syncthreads();

    const int n1i = chunk * 256 + tid;
    const size_t row = (size_t)b * N1_ + n1i;
    const float px = xyz1[row * 3 + 0];
    const float py = xyz1[row * 3 + 1];
    const float pz = xyz1[row * 3 + 2];

    float s0 = 3.0e38f, s1 = 3.0e38f, s2 = 3.0e38f;
    int   i0 = 0, i1 = 0, i2 = 0;
    #pragma unroll 4
    for (int j = 0; j < N2_; ++j) {
        const float dx = px - s_xyz[3 * j + 0];
        const float dy = py - s_xyz[3 * j + 1];
        const float dz = pz - s_xyz[3 * j + 2];
        const float sq = dx * dx + dy * dy + dz * dz;
        if (sq < s0)      { s2 = s1; i2 = i1; s1 = s0; i1 = i0; s0 = sq; i0 = j; }
        else if (sq < s1) { s2 = s1; i2 = i1; s1 = sq; i1 = j; }
        else if (sq < s2) { s2 = sq; i2 = j; }
    }

    const float d0 = sqrtf(fmaxf(s0, 0.0f));
    const float d1 = sqrtf(fmaxf(s1, 0.0f));
    const float d2 = sqrtf(fmaxf(s2, 0.0f));
    float w0 = 1.0f / (d0 + 1e-8f);
    float w1 = 1.0f / (d1 + 1e-8f);
    float w2 = 1.0f / (d2 + 1e-8f);
    const float inv = 1.0f / (w0 + w1 + w2);
    w0 *= inv; w1 *= inv; w2 *= inv;

    const float4* p0 = (const float4*)(points2 + ((size_t)b * N2_ + i0) * C2_);
    const float4* p1 = (const float4*)(points2 + ((size_t)b * N2_ + i1) * C2_);
    const float4* p2 = (const float4*)(points2 + ((size_t)b * N2_ + i2) * C2_);
    uint2* dst = (uint2*)(A0 + row * K0_);     // 1 uint2 = 4 bf16
    #pragma unroll 4
    for (int c = 0; c < C2_ / 4; ++c) {
        const float4 a = p0[c], bb = p1[c], cc = p2[c];
        float4 o;
        o.x = w0 * a.x + w1 * bb.x + w2 * cc.x;
        o.y = w0 * a.y + w1 * bb.y + w2 * cc.y;
        o.z = w0 * a.z + w1 * bb.z + w2 * cc.z;
        o.w = w0 * a.w + w1 * bb.w + w2 * cc.w;
        dst[c] = make_uint2(pack_bf16x2(o.x, o.y), pack_bf16x2(o.z, o.w));
    }
    const float4* pp1 = (const float4*)(points1 + row * C1_);
    #pragma unroll 4
    for (int c = 0; c < C1_ / 4; ++c) {
        const float4 v = pp1[c];
        dst[C2_ / 4 + c] = make_uint2(pack_bf16x2(v.x, v.y), pack_bf16x2(v.z, v.w));
    }
}

// ---------------------------------------------------------------------------
// bf16 WMMA GEMM:  C[r][o] = sum_k A[r][k] * W[o][k]   (fp32 accumulate)
// A: R x K bf16 row-major; W: 256 x K bf16 row-major; C: R x 256 fp32.
// Block = 256 threads = 8 waves; block tile = 32 rows x 256 cols.
// Wave tile = 16 rows x 64 cols (4 accumulators share one A fragment).
// v_wmma_f32_16x16x32_bf16, K stepped by 32.
// Fragment layout per ISA 7.12.2 (16-bit A 16x32 / B 32x16, wave32):
//   A lane: row=l16, K = 8*half+{0..7} and 16+8*half+{0..7}  -> two b128 loads
//   B lane: col=l16, K = 8*half+{0..7} and 16+8*half+{0..7}  -> two b128 loads
// Also accumulates per-channel sum / sum^2 for training-mode BatchNorm.
// ---------------------------------------------------------------------------
__global__ void gemm_bf16_wmma_kernel(const uint16_t* __restrict__ A,
                                      const uint16_t* __restrict__ W,
                                      float* __restrict__ C,
                                      float* __restrict__ statsSum,
                                      float* __restrict__ statsSq,
                                      int K) {
    __shared__ float lsum[256];
    __shared__ float lsq[256];
    const int tid = threadIdx.x;
    lsum[tid] = 0.0f;
    lsq[tid]  = 0.0f;
    __syncthreads();

    const int wave  = tid >> 5;
    const int lane  = tid & 31;
    const int half  = lane >> 4;
    const int l16   = lane & 15;
    const int waveM = wave & 1;
    const int waveN = wave >> 1;
    const int m0 = blockIdx.x * 32 + waveM * 16;
    const int n0 = waveN * 64;

    union Frag { v16bf v; uint4 u[2]; };

    v8f acc[4];
    #pragma unroll
    for (int t = 0; t < 4; ++t)
        #pragma unroll
        for (int v = 0; v < 8; ++v) acc[t][v] = 0.0f;

    const uint16_t* Arow = A + (size_t)(m0 + l16) * K;
    for (int k = 0; k < K; k += 32) {
        const int ka = k + 8 * half;
        Frag a;
        a.u[0] = *(const uint4*)(Arow + ka);
        a.u[1] = *(const uint4*)(Arow + ka + 16);
        #pragma unroll
        for (int t = 0; t < 4; ++t) {
            const int col = n0 + t * 16 + l16;
            const uint16_t* Bp = W + (size_t)col * K + k + 8 * half;
            Frag bf;
            bf.u[0] = *(const uint4*)(Bp);
            bf.u[1] = *(const uint4*)(Bp + 16);
            acc[t] = __builtin_amdgcn_wmma_f32_16x16x32_bf16(
                false, a.v, false, bf.v, (short)0, acc[t], false, false);
        }
    }

    // epilogue: store raw pre-BN activations (fp32) + accumulate channel stats
    #pragma unroll
    for (int t = 0; t < 4; ++t) {
        const int col = n0 + t * 16 + l16;
        float s = 0.0f, q = 0.0f;
        #pragma unroll
        for (int v = 0; v < 8; ++v) {
            const float x = acc[t][v];
            const int row = m0 + v + 8 * half;
            C[(size_t)row * 256 + col] = x;
            s += x; q += x * x;
        }
        atomicAdd(&lsum[col], s);
        atomicAdd(&lsq[col], q);
    }
    __syncthreads();
    atomicAdd(&statsSum[tid], lsum[tid]);
    atomicAdd(&statsSq[tid],  lsq[tid]);
}

// ---------------------------------------------------------------------------
// Finalize BN: scale = gamma*rsqrt(var+eps), shift = beta - mean*scale
// ---------------------------------------------------------------------------
__global__ void finalize_bn_kernel(const float* __restrict__ sum,
                                   const float* __restrict__ sq,
                                   const float* __restrict__ gamma,
                                   const float* __restrict__ beta,
                                   float* __restrict__ scale,
                                   float* __restrict__ shift,
                                   float invR) {
    const int c = threadIdx.x;
    const float m   = sum[c] * invR;
    const float var = sq[c] * invR - m * m;
    const float sc  = gamma[c] * rsqrtf(var + BN_EPS);
    scale[c] = sc;
    shift[c] = beta[c] - m * sc;
}

// ---------------------------------------------------------------------------
// BN0 + ReLU on fp32 h0, convert to bf16 A1 (input of GEMM1)
// ---------------------------------------------------------------------------
__global__ void bnrelu_cvt_kernel(const float* __restrict__ h0,
                                  const float* __restrict__ scale,
                                  const float* __restrict__ shift,
                                  uint32_t* __restrict__ A1) {
    const size_t idx = (size_t)blockIdx.x * 256 + threadIdx.x; // float4 index
    const int c = ((int)(idx & 63)) * 4;
    float4 v = ((const float4*)h0)[idx];
    v.x = fmaxf(v.x * scale[c + 0] + shift[c + 0], 0.0f);
    v.y = fmaxf(v.y * scale[c + 1] + shift[c + 1], 0.0f);
    v.z = fmaxf(v.z * scale[c + 2] + shift[c + 2], 0.0f);
    v.w = fmaxf(v.w * scale[c + 3] + shift[c + 3], 0.0f);
    ((uint2*)A1)[idx] = make_uint2(pack_bf16x2(v.x, v.y), pack_bf16x2(v.z, v.w));
}

// ---------------------------------------------------------------------------
// Final in-place BN1 + ReLU over d_out (R_ x 256 fp32)
// ---------------------------------------------------------------------------
__global__ void bn_relu_out_kernel(float* __restrict__ out,
                                   const float* __restrict__ scale,
                                   const float* __restrict__ shift) {
    const size_t idx = (size_t)blockIdx.x * 256 + threadIdx.x; // float4 index
    const int c = ((int)(idx & 63)) * 4;
    float4 v = ((float4*)out)[idx];
    v.x = fmaxf(v.x * scale[c + 0] + shift[c + 0], 0.0f);
    v.y = fmaxf(v.y * scale[c + 1] + shift[c + 1], 0.0f);
    v.z = fmaxf(v.z * scale[c + 2] + shift[c + 2], 0.0f);
    v.w = fmaxf(v.w * scale[c + 3] + shift[c + 3], 0.0f);
    ((float4*)out)[idx] = v;
}

// ---------------------------------------------------------------------------
extern "C" void kernel_launch(void* const* d_in, const int* in_sizes, int n_in,
                              void* d_out, int out_size, void* d_ws, size_t ws_size,
                              hipStream_t stream) {
    const float* xyz1    = (const float*)d_in[0];
    const float* xyz2    = (const float*)d_in[1];
    const float* points1 = (const float*)d_in[2];
    const float* points2 = (const float*)d_in[3];
    const float* W0      = (const float*)d_in[4];
    const float* gamma0  = (const float*)d_in[5];
    const float* beta0   = (const float*)d_in[6];
    const float* W1      = (const float*)d_in[7];
    const float* gamma1  = (const float*)d_in[8];
    const float* beta1   = (const float*)d_in[9];
    float* out = (float*)d_out;

    // workspace layout (all offsets 16B aligned)
    char* w = (char*)d_ws;
    uint16_t* A0  = (uint16_t*)(w);                      // R*384 bf16  = 50,331,648 B
    float*    h0  = (float*)   (w + 50331648);           // R*256 f32   = 67,108,864 B
    uint16_t* A1  = (uint16_t*)(w + 117440512);          // R*256 bf16  = 33,554,432 B
    uint16_t* W0b = (uint16_t*)(w + 150994944);          // 256*384 bf16=    196,608 B
    uint16_t* W1b = (uint16_t*)(w + 151191552);          // 256*256 bf16=    131,072 B
    float* stats  = (float*)   (w + 151322624);          // 2048 floats
    float* sum0 = stats;          float* sq0    = stats + 256;
    float* sum1 = stats + 512;    float* sq1    = stats + 768;
    float* scale0 = stats + 1024; float* shift0 = stats + 1280;
    float* scale1 = stats + 1536; float* shift1 = stats + 1792;

    zero_stats_kernel<<<4, 256, 0, stream>>>(stats);

    cvt_f32_bf16_kernel<<<(M0_ * K0_ / 2) / 256, 256, 0, stream>>>(W0, (uint32_t*)W0b);
    cvt_f32_bf16_kernel<<<(M1_ * M0_ / 2) / 256, 256, 0, stream>>>(W1, (uint32_t*)W1b);

    knn_interp_kernel<<<B_ * (N1_ / 256), 256, 0, stream>>>(
        xyz1, xyz2, points1, points2, A0);

    gemm_bf16_wmma_kernel<<<R_ / 32, 256, 0, stream>>>(
        A0, W0b, h0, sum0, sq0, K0_);

    finalize_bn_kernel<<<1, 256, 0, stream>>>(
        sum0, sq0, gamma0, beta0, scale0, shift0, 1.0f / (float)R_);

    bnrelu_cvt_kernel<<<(R_ * M0_ / 4) / 256, 256, 0, stream>>>(
        h0, scale0, shift0, (uint32_t*)A1);

    gemm_bf16_wmma_kernel<<<R_ / 32, 256, 0, stream>>>(
        A1, W1b, out, sum1, sq1, M0_);

    finalize_bn_kernel<<<1, 256, 0, stream>>>(
        sum1, sq1, gamma1, beta1, scale1, shift1, 1.0f / (float)R_);

    bn_relu_out_kernel<<<(R_ * M1_ / 4) / 256, 256, 0, stream>>>(
        out, scale1, shift1);
}
